// MyModel_20074677141543
// MI455X (gfx1250) — matrix operand order repaired and checked by reading, not compiled, not run on hardware
//
#include <hip/hip_runtime.h>
#include <hip/hip_bf16.h>

typedef __attribute__((ext_vector_type(16))) _Float16 v16h;
typedef __attribute__((ext_vector_type(8)))  _Float16 v8h;
typedef __attribute__((ext_vector_type(8)))  float    v8f;
typedef __attribute__((ext_vector_type(4)))  float    v4f;

#define N_ROWS 65536
#define P_DIM  512
#define HD_DIM 256
#define CIN    768          // 512 + 256
#define H1     224          // 200 padded to 32-multiple
#define H2     128          // 100 padded
#define H3     128          // 100 padded
#define ROWS_PER_BLOCK 64
#define MLP_BLOCKS (N_ROWS / ROWS_PER_BLOCK)   // 1024
#define HSTAT_BLOCKS 64
#define HSTAT_ROWS  (N_ROWS / HSTAT_BLOCKS)    // 1024

// ---- workspace layout (bytes) ----
#define W1P_OFF    0u          // 224*768 f16 = 344064 B
#define W12P_OFF   344064u     // 128*224 f16 = 57344 B
#define W13P_OFF   401408u     // 128*128 f16 = 32768 B
#define W2P_OFF    434176u     // 512*128 f16 = 131072 B
#define B1P_OFF    565248u     // 224 f32
#define B12P_OFF   566144u     // 128 f32
#define B13P_OFF   566656u     // 128 f32
#define P0_OFF     567168u     // 1024 f32 (per-mlp-block part0 partials)
#define COLSUM_OFF 571264u     // 64*256 f32
#define SQ_OFF     636800u     // 64 f32
#define DIFF_OFF   637056u     // 64 f32

union FragH { v16h v; _Float16 e[16]; };

// ---------------- fragment loaders (wave32, 16x16x32 f16 WMMA layouts) -------------

// A fragment: lane m = lane&15; lanes<16 hold K = {kc*32+0..7, kc*32+16..23},
// lanes>=16 hold K = {kc*32+8..15, kc*32+24..31}. Source: f32 concat(input, h).
__device__ __forceinline__ v16h load_a_concat(const float* __restrict__ inp,
                                              const float* __restrict__ hp,
                                              int row, int kc, int lane) {
    int s  = (lane & 16) ? 8 : 0;
    int f0 = kc * 32 + s;
    int f1 = f0 + 16;
    const float* p0 = (f0 < P_DIM) ? (inp + (size_t)row * P_DIM + f0)
                                   : (hp  + (size_t)row * HD_DIM + (f0 - P_DIM));
    const float* p1 = (f1 < P_DIM) ? (inp + (size_t)row * P_DIM + f1)
                                   : (hp  + (size_t)row * HD_DIM + (f1 - P_DIM));
    v4f a0 = *(const v4f*)p0;
    v4f a1 = *(const v4f*)(p0 + 4);
    v4f b0 = *(const v4f*)p1;
    v4f b1 = *(const v4f*)(p1 + 4);
    FragH r;
#pragma unroll
    for (int i = 0; i < 4; ++i) {
        r.e[i]      = (_Float16)a0[i];
        r.e[4 + i]  = (_Float16)a1[i];
        r.e[8 + i]  = (_Float16)b0[i];
        r.e[12 + i] = (_Float16)b1[i];
    }
    return r.v;
}

// A fragment from per-wave LDS activation tile (row-major [16][Kpad] f16).
__device__ __forceinline__ v16h load_a_lds(const _Float16* __restrict__ act,
                                           int Kpad, int kc, int lane) {
    int m = lane & 15;
    int s = (lane & 16) ? 8 : 0;
    const _Float16* p = act + m * Kpad + kc * 32 + s;
    v8h lo = *(const v8h*)p;          // K = kc*32+s .. +7
    v8h hi = *(const v8h*)(p + 16);   // K = kc*32+16+s .. +7
    FragH r;
#pragma unroll
    for (int i = 0; i < 8; ++i) { r.e[i] = lo[i]; r.e[8 + i] = hi[i]; }
    return r.v;
}

// B fragment: col = jbase + (lane&15); lanes<16 hold K = {kc*32+0..7, kc*32+16..23},
// lanes>=16 hold K = {kc*32+8..15, kc*32+24..31}. Weights are f16 row-major [out][Kpad]
// (i.e. B = W^T accessed column-major == W row-major, contiguous 32B run).
__device__ __forceinline__ v16h load_b16(const _Float16* __restrict__ Wp,
                                         int Kpad, int jbase, int kc, int lane) {
    int col = jbase + (lane & 15);
    int k0  = kc * 32 + ((lane & 16) ? 8 : 0);
    const _Float16* p = Wp + (size_t)col * Kpad + k0;
    v8h lo = *(const v8h*)p;
    v8h hi = *(const v8h*)(p + 16);
    FragH r;
#pragma unroll
    for (int i = 0; i < 8; ++i) { r.e[i] = lo[i]; r.e[8 + i] = hi[i]; }
    return r.v;
}

// ---------------- kernel 1: pad + convert weights to f16 ----------------
__global__ __launch_bounds__(256) void prep_kernel(
    const float* __restrict__ W1,  const float* __restrict__ b1,
    const float* __restrict__ W12, const float* __restrict__ b12,
    const float* __restrict__ W13, const float* __restrict__ b13,
    const float* __restrict__ W2,
    _Float16* __restrict__ W1p, _Float16* __restrict__ W12p,
    _Float16* __restrict__ W13p, _Float16* __restrict__ W2p,
    float* __restrict__ b1p, float* __restrict__ b12p, float* __restrict__ b13p) {
    int i = blockIdx.x * blockDim.x + threadIdx.x;
    if (i < H1 * CIN) {
        int o = i / CIN, k = i % CIN;
        W1p[i] = (_Float16)((o < 200) ? W1[o * CIN + k] : 0.f);
    }
    if (i < H2 * H1) {
        int o = i / H1, k = i % H1;
        W12p[i] = (_Float16)((o < 100 && k < 200) ? W12[o * 200 + k] : 0.f);
    }
    if (i < H3 * H2) {
        int o = i / H2, k = i % H2;
        W13p[i] = (_Float16)((o < 100 && k < 100) ? W13[o * 100 + k] : 0.f);
    }
    if (i < P_DIM * H3) {
        int o = i / H3, k = i % H3;
        W2p[i] = (_Float16)((k < 100) ? W2[o * 100 + k] : 0.f);
    }
    if (i < H1) b1p[i]  = (i < 200) ? b1[i]  : 0.f;
    if (i < H2) b12p[i] = (i < 100) ? b12[i] : 0.f;
    if (i < H3) b13p[i] = (i < 100) ? b13[i] : 0.f;
}

// ---------------- kernel 2: fused 4-layer WMMA MLP + part0 partials ----------------
__global__ __launch_bounds__(128) void mlp_kernel(
    const float* __restrict__ inp, const float* __restrict__ hp,
    const float* __restrict__ b2,
    const _Float16* __restrict__ W1p, const _Float16* __restrict__ W12p,
    const _Float16* __restrict__ W13p, const _Float16* __restrict__ W2p,
    const float* __restrict__ b1p, const float* __restrict__ b12p,
    const float* __restrict__ b13p,
    float* __restrict__ out, float* __restrict__ p0_partial) {
    __shared__ _Float16 s_act1[4][16 * H1];
    __shared__ _Float16 s_act2[4][16 * H2];
    __shared__ _Float16 s_act3[4][16 * H3];
    __shared__ float    s_red[128];

    const int tid   = threadIdx.x;
    const int wave  = tid >> 5;         // wave32
    const int lane  = tid & 31;
    const int col16 = lane & 15;
    const int mrow  = (lane >> 4) * 8;  // D-fragment row base for this lane half
    const int rowbase = blockIdx.x * ROWS_PER_BLOCK + wave * 16;
    const int arow    = rowbase + col16; // A-fragment source row for this lane

    // ---- layer 1: [16 x 768] x [768 x 224] ----
    for (int j = 0; j < H1 / 16; ++j) {
        float bv = b1p[j * 16 + col16];
        v8f c = {bv, bv, bv, bv, bv, bv, bv, bv};
#pragma unroll 4
        for (int kc = 0; kc < CIN / 32; ++kc) {
            v16h a = load_a_concat(inp, hp, arow, kc, lane);
            v16h b = load_b16(W1p, CIN, j * 16, kc, lane);
            c = __builtin_amdgcn_wmma_f32_16x16x32_f16(false, a, false, b,
                                                       (short)0, c, false, false);
        }
#pragma unroll
        for (int v = 0; v < 8; ++v) {
            float x = c[v];
            x = x > 0.f ? x : 0.f;
            s_act1[wave][(mrow + v) * H1 + j * 16 + col16] = (_Float16)x;
        }
    }

    // ---- layer 2: [16 x 224] x [224 x 128] ----
    for (int j = 0; j < H2 / 16; ++j) {
        float bv = b12p[j * 16 + col16];
        v8f c = {bv, bv, bv, bv, bv, bv, bv, bv};
#pragma unroll
        for (int kc = 0; kc < H1 / 32; ++kc) {
            v16h a = load_a_lds(s_act1[wave], H1, kc, lane);
            v16h b = load_b16(W12p, H1, j * 16, kc, lane);
            c = __builtin_amdgcn_wmma_f32_16x16x32_f16(false, a, false, b,
                                                       (short)0, c, false, false);
        }
#pragma unroll
        for (int v = 0; v < 8; ++v) {
            float x = c[v];
            x = x > 0.f ? x : 0.f;
            s_act2[wave][(mrow + v) * H2 + j * 16 + col16] = (_Float16)x;
        }
    }

    // ---- layer 3: [16 x 128] x [128 x 128] ----
    for (int j = 0; j < H3 / 16; ++j) {
        float bv = b13p[j * 16 + col16];
        v8f c = {bv, bv, bv, bv, bv, bv, bv, bv};
#pragma unroll
        for (int kc = 0; kc < H2 / 32; ++kc) {
            v16h a = load_a_lds(s_act2[wave], H2, kc, lane);
            v16h b = load_b16(W13p, H2, j * 16, kc, lane);
            c = __builtin_amdgcn_wmma_f32_16x16x32_f16(false, a, false, b,
                                                       (short)0, c, false, false);
        }
#pragma unroll
        for (int v = 0; v < 8; ++v) {
            float x = c[v];
            x = x > 0.f ? x : 0.f;
            s_act3[wave][(mrow + v) * H3 + j * 16 + col16] = (_Float16)x;
        }
    }

    // ---- layer 4: [16 x 128] x [128 x 512], fused part0 accumulation ----
    float p0acc = 0.f;
    for (int j = 0; j < P_DIM / 16; ++j) {
        float bv = b2[j * 16 + col16];
        v8f c = {bv, bv, bv, bv, bv, bv, bv, bv};
#pragma unroll
        for (int kc = 0; kc < H3 / 32; ++kc) {
            v16h a = load_a_lds(s_act3[wave], H3, kc, lane);
            v16h b = load_b16(W2p, H3, j * 16, kc, lane);
            c = __builtin_amdgcn_wmma_f32_16x16x32_f16(false, a, false, b,
                                                       (short)0, c, false, false);
        }
        int col = j * 16 + col16;
#pragma unroll
        for (int v = 0; v < 8; ++v) {
            int row = rowbase + mrow + v;
            float val = c[v];
            out[(size_t)row * P_DIM + col] = val;
            if (row < N_ROWS - 1) {
                float d = val - inp[(size_t)(row + 1) * P_DIM + col];
                p0acc += d * d;
            }
        }
    }

    // deterministic block reduction of part0 partial
    s_red[tid] = p0acc;
    __syncthreads();
    if (tid == 0) {
        float s = 0.f;
        for (int i = 0; i < 128; ++i) s += s_red[i];
        p0_partial[blockIdx.x] = s;
    }
}

// ---------------- kernel 3: h statistics (part1/2/3 partials) ----------------
__global__ __launch_bounds__(256) void hstats_kernel(
    const float* __restrict__ hp,
    float* __restrict__ colsum, float* __restrict__ sqp, float* __restrict__ diffp) {
    __shared__ float s_sq[256];
    __shared__ float s_df[256];
    int t  = threadIdx.x;
    int r0 = blockIdx.x * HSTAT_ROWS;
    float prev = hp[(size_t)r0 * HD_DIM + t];
    float cs = prev, sq = prev * prev, df = 0.f;
    for (int r = r0 + 1; r < r0 + HSTAT_ROWS; ++r) {
        float cur = hp[(size_t)r * HD_DIM + t];
        df += fabsf(cur - prev);
        cs += cur;
        sq += cur * cur;
        prev = cur;
    }
    int rn = r0 + HSTAT_ROWS;
    if (rn < N_ROWS) df += fabsf(hp[(size_t)rn * HD_DIM + t] - prev);
    colsum[blockIdx.x * HD_DIM + t] = cs;
    s_sq[t] = sq;
    s_df[t] = df;
    __syncthreads();
    if (t == 0) {
        float a = 0.f, b = 0.f;
        for (int i = 0; i < 256; ++i) { a += s_sq[i]; b += s_df[i]; }
        sqp[blockIdx.x]   = a;
        diffp[blockIdx.x] = b;
    }
}

// ---------------- kernel 4: finalize scalars ----------------
__global__ __launch_bounds__(256) void finalize_kernel(
    const float* __restrict__ colsum, const float* __restrict__ sqp,
    const float* __restrict__ diffp,  const float* __restrict__ p0p,
    float* __restrict__ outs) {
    __shared__ float s_abs[256];
    int t = threadIdx.x;
    float cs = 0.f;
    for (int b = 0; b < HSTAT_BLOCKS; ++b) cs += colsum[b * HD_DIM + t];
    s_abs[t] = fabsf(cs / (float)N_ROWS);
    __syncthreads();
    if (t == 0) {
        float p1 = 0.f;
        for (int i = 0; i < 256; ++i) p1 += s_abs[i];
        p1 /= (float)HD_DIM;
        float S = 0.f;
        for (int b = 0; b < HSTAT_BLOCKS; ++b) S += sqp[b];
        S /= (float)N_ROWS;
        float p2 = fabsf(S / (float)HD_DIM - 1.f);
        float d = 0.f;
        for (int b = 0; b < HSTAT_BLOCKS; ++b) d += diffp[b];
        float p3 = d / (float)(N_ROWS - 1) / (float)HD_DIM;
        float q = 0.f;
        for (int b = 0; b < MLP_BLOCKS; ++b) q += p0p[b];
        float p0 = sqrtf(q) / (float)(N_ROWS - 1);
        outs[0] = p0;
        outs[1] = p1;
        outs[2] = p2;
        outs[3] = p3;
    }
}

extern "C" void kernel_launch(void* const* d_in, const int* in_sizes, int n_in,
                              void* d_out, int out_size, void* d_ws, size_t ws_size,
                              hipStream_t stream) {
    (void)in_sizes; (void)n_in; (void)out_size; (void)ws_size;
    const float* inp = (const float*)d_in[0];
    const float* hp  = (const float*)d_in[1];
    const float* W1  = (const float*)d_in[2];
    const float* b1  = (const float*)d_in[3];
    const float* W12 = (const float*)d_in[4];
    const float* b12 = (const float*)d_in[5];
    const float* W13 = (const float*)d_in[6];
    const float* b13 = (const float*)d_in[7];
    const float* W2  = (const float*)d_in[8];
    const float* b2  = (const float*)d_in[9];
    // d_in[10], d_in[11]: idx0=0, idx1=N (full range)

    char* ws = (char*)d_ws;
    _Float16* W1p  = (_Float16*)(ws + W1P_OFF);
    _Float16* W12p = (_Float16*)(ws + W12P_OFF);
    _Float16* W13p = (_Float16*)(ws + W13P_OFF);
    _Float16* W2p  = (_Float16*)(ws + W2P_OFF);
    float* b1p  = (float*)(ws + B1P_OFF);
    float* b12p = (float*)(ws + B12P_OFF);
    float* b13p = (float*)(ws + B13P_OFF);
    float* p0p  = (float*)(ws + P0_OFF);
    float* cso  = (float*)(ws + COLSUM_OFF);
    float* sqp  = (float*)(ws + SQ_OFF);
    float* dfp  = (float*)(ws + DIFF_OFF);

    float* out = (float*)d_out;
    float* out_scalars = out + (size_t)N_ROWS * P_DIM;

    int prep_blocks = (H1 * CIN + 255) / 256;  // 672, covers all sections
    prep_kernel<<<prep_blocks, 256, 0, stream>>>(W1, b1, W12, b12, W13, b13, W2,
                                                 W1p, W12p, W13p, W2p,
                                                 b1p, b12p, b13p);
    mlp_kernel<<<MLP_BLOCKS, 128, 0, stream>>>(inp, hp, b2,
                                               W1p, W12p, W13p, W2p,
                                               b1p, b12p, b13p,
                                               out, p0p);
    hstats_kernel<<<HSTAT_BLOCKS, 256, 0, stream>>>(hp, cso, sqp, dfp);
    finalize_kernel<<<1, 256, 0, stream>>>(cso, sqp, dfp, p0p, out_scalars);
}
